// CNP_Net_89283780149635
// MI455X (gfx1250) — matrix operand order repaired and checked by reading, not compiled, not run on hardware
//
#include <hip/hip_runtime.h>
#include <hip/hip_bf16.h>
#include <math.h>

typedef __attribute__((ext_vector_type(16))) _Float16 v16h;
typedef __attribute__((ext_vector_type(8)))  _Float16 v8h;
typedef __attribute__((ext_vector_type(8)))  float    v8f;

__device__ __forceinline__ v8f wmma_f16f32(v16h a, v16h b, v8f c) {
  // D = A(16x32 f16) * B(32x16 f16) + C(16x16 f32)
  return __builtin_amdgcn_wmma_f32_16x16x32_f16(false, a, false, b, (short)0, c, false, false);
}

// Build a B-fragment (32x16 logical, f16) from a row-major f32 weight matrix.
// B layout (wave32): lane holds column n = lane&15; VGPR v holds K rows {kb+2v,kb+2v+1}
// (+8 for v>=4), kb = 0 for lanes 0-15 and 8 for lanes 16-31. Out-of-range
// K/N (padding) are zero-filled.
__device__ __forceinline__ v16h make_bfrag(const float* __restrict__ W, int stride,
                                           int K0, int N0, int kRows, int nCols, int lane) {
  const int n = (lane & 15) + N0;
  const int kbase = K0 + ((lane & 16) ? 8 : 0);
  v16h b;
#pragma unroll
  for (int v = 0; v < 8; ++v) {
    const int k0 = kbase + 2 * v + (v >= 4 ? 8 : 0);
    const int k1 = k0 + 1;
    const float w0 = (k0 < kRows && n < nCols) ? W[k0 * stride + n] : 0.0f;
    const float w1 = (k1 < kRows && n < nCols) ? W[k1 * stride + n] : 0.0f;
    b[2 * v]     = (_Float16)w0;
    b[2 * v + 1] = (_Float16)w1;
  }
  return b;
}

// A layout (16x32 f16): lane holds row m = lane&15; halves 0-7 = K(hi..hi+7),
// halves 8-15 = K(16+hi..16+hi+7), hi = (lane<16 ? 0 : 8). Tile is row-major in LDS.
__device__ __forceinline__ v16h lds_afrag(const _Float16* tile, int stride, int K0, int lane) {
  const int m = lane & 15;
  const int hi = (lane & 16) ? 8 : 0;
  const _Float16* p0 = tile + m * stride + K0 + hi;
  const v8h lo = *(const v8h*)p0;
  const v8h hh = *(const v8h*)(p0 + 16);
  v16h a;
#pragma unroll
  for (int j = 0; j < 8; ++j) { a[j] = lo[j]; a[j + 8] = hh[j]; }
  return a;
}

// Same but logical K is only 16 wide: upper 8 halves (K 16..31) forced to zero.
__device__ __forceinline__ v16h lds_afrag_k16(const _Float16* tile, int stride, int lane) {
  const int m = lane & 15;
  const int hi = (lane & 16) ? 8 : 0;
  const v8h lo = *(const v8h*)(tile + m * stride + hi);
  v16h a;
#pragma unroll
  for (int j = 0; j < 8; ++j) { a[j] = lo[j]; a[j + 8] = (_Float16)0.0f; }
  return a;
}

// C layout (16x16 f32): lane holds column n = lane&15; VGPR r holds row
// m = r + (lane<16 ? 0 : 8). Store relu(C) as f16 into a row-major LDS tile.
__device__ __forceinline__ void store_relu_half(_Float16* tile, int stride, int N0,
                                                v8f c, int lane) {
  const int n = (lane & 15) + N0;
  const int mb = (lane & 16) ? 8 : 0;
#pragma unroll
  for (int r = 0; r < 8; ++r) {
    float u = c[r];
    u = u > 0.0f ? u : 0.0f;
    tile[(mb + r) * stride + n] = (_Float16)u;
  }
}

// ---------------------------------------------------------------------------
// Kernel 0: zero the workspace accumulators (h1 column sums [0..31], lp [32]).
__global__ void init_ws_kernel(float* __restrict__ ws) {
  const int i = threadIdx.x;
  if (i < 33) ws[i] = 0.0f;
}

// ---------------------------------------------------------------------------
// Kernel 1: encoder. Per 16-row tile: h0 = relu(O@Wh0+bh0) per lane (2->8),
// then 8->32 via two WMMA (K padded 8->32), accumulate column sums of relu(h1).
__global__ void __launch_bounds__(32) enc_kernel(
    const float* __restrict__ O, const float* __restrict__ Wh0,
    const float* __restrict__ bh0, const float* __restrict__ Wh1,
    const float* __restrict__ bh1, float* __restrict__ ws_sum, int n) {
  const int lane = threadIdx.x;
  __shared__ float ssum[32];
  ssum[lane] = 0.0f;
  __syncthreads();

  float w0a[8], w0b[8], b0[8];
#pragma unroll
  for (int j = 0; j < 8; ++j) { w0a[j] = Wh0[j]; w0b[j] = Wh0[8 + j]; b0[j] = bh0[j]; }
  const v16h B0 = make_bfrag(Wh1, 32, 0, 0, 8, 32, lane);
  const v16h B1 = make_bfrag(Wh1, 32, 0, 16, 8, 32, lane);
  const float bias_lo = bh1[lane & 15];
  const float bias_hi = bh1[16 + (lane & 15)];

  float p0 = 0.0f, p1 = 0.0f;
  const int ntiles = (n + 15) >> 4;
  for (int t = blockIdx.x; t < ntiles; t += gridDim.x) {
    const int row = t * 16 + (lane & 15);
    const int rc = row < n ? row : n - 1;
    const float o0 = O[2 * rc];
    const float o1 = O[2 * rc + 1];
    v16h a;
#pragma unroll
    for (int j = 0; j < 16; ++j) a[j] = (_Float16)0.0f;
    if (lane < 16) {  // lanes 16-31 hold padded K rows 8..15 -> zero
#pragma unroll
      for (int j = 0; j < 8; ++j) {
        const float h = fmaf(o1, w0b[j], fmaf(o0, w0a[j], b0[j]));
        a[j] = (_Float16)(h > 0.0f ? h : 0.0f);
      }
    }
    v8f c0, c1;
#pragma unroll
    for (int r = 0; r < 8; ++r) { c0[r] = bias_lo; c1[r] = bias_hi; }
    c0 = wmma_f16f32(a, B0, c0);
    c1 = wmma_f16f32(a, B1, c1);
    const int mb = (lane & 16) ? 8 : 0;
    if (t * 16 + 16 <= n) {
      // uniform full-tile fast path: no per-row masking, no exec churn
#pragma unroll
      for (int r = 0; r < 8; ++r) {
        float u0 = c0[r]; u0 = u0 > 0.0f ? u0 : 0.0f;
        float u1 = c1[r]; u1 = u1 > 0.0f ? u1 : 0.0f;
        p0 += u0;
        p1 += u1;
      }
    } else {
      // ragged last tile: branch-free per-row select
#pragma unroll
      for (int r = 0; r < 8; ++r) {
        const float sel = (t * 16 + mb + r < n) ? 1.0f : 0.0f;
        float u0 = c0[r]; u0 = u0 > 0.0f ? u0 : 0.0f;
        float u1 = c1[r]; u1 = u1 > 0.0f ? u1 : 0.0f;
        p0 = fmaf(sel, u0, p0);
        p1 = fmaf(sel, u1, p1);
      }
    }
  }
  atomicAdd(&ssum[lane & 15], p0);
  atomicAdd(&ssum[16 + (lane & 15)], p1);
  __syncthreads();
  atomicAdd(&ws_sum[lane], ssum[lane]);
}

// ---------------------------------------------------------------------------
// Kernel 2: r = mean(h1)@Wh2 + bh2 (128), then c = r@Wg0[:128] + bg0 (64) -> ws[64..127]
__global__ void __launch_bounds__(128) combine_kernel(
    const float* __restrict__ Wh2, const float* __restrict__ bh2,
    const float* __restrict__ Wg0, const float* __restrict__ bg0,
    float* __restrict__ ws, int nctx) {
  __shared__ float r[128];
  const int t = threadIdx.x;
  const float inv = 1.0f / (float)nctx;
  {
    float s = bh2[t];
#pragma unroll
    for (int k = 0; k < 32; ++k) s = fmaf(ws[k] * inv, Wh2[k * 128 + t], s);
    r[t] = s;
  }
  __syncthreads();
  if (t < 64) {
    float s = bg0[t];
#pragma unroll 8
    for (int j = 0; j < 128; ++j) s = fmaf(r[j], Wg0[j * 64 + t], s);
    ws[64 + t] = s;
  }
}

// ---------------------------------------------------------------------------
// Kernel 3: decoder. g0 = relu(c + t0*Wg0[128,:]) built analytically as A frags,
// then 64->32 (4 WMMA), 32->16 (1 WMMA), 16->8 and 8->2 (padded WMMA). Writes phi
// (coalesced float2 per row) and accumulates Gaussian log-prob partial sums.
__global__ void __launch_bounds__(32) dec_kernel(
    const float* __restrict__ T, const float* __restrict__ Wg0,
    const float* __restrict__ cvec,
    const float* __restrict__ Wg1, const float* __restrict__ bg1,
    const float* __restrict__ Wg2, const float* __restrict__ bg2,
    const float* __restrict__ Wg3, const float* __restrict__ bg3,
    const float* __restrict__ Wg4, const float* __restrict__ bg4,
    float* __restrict__ out, float* __restrict__ lp_accum, int n) {
  const int lane = threadIdx.x;
  __shared__ __align__(16) _Float16 t1[16 * 40];  // g1: 16x32, stride 40 (16B aligned rows)
  __shared__ __align__(16) _Float16 t2[16 * 24];  // g2: 16x16, stride 24
  __shared__ __align__(16) _Float16 t3[16 * 24];  // g3: 16x16 (cols>=8 are zero)
  __shared__ float phis[32];
  __shared__ float red[32];

  const int nlo = lane & 15;
  const int hi8 = (lane & 16) ? 8 : 0;

  // Per-lane layer-0 coefficients: the K indices this lane owns in A layout.
  float cA[32], wA[32];
  const float* wl = Wg0 + 128 * 64;  // last row of Wg0: multiplies t0
#pragma unroll
  for (int g = 0; g < 4; ++g) {
    const int base = g * 16 + hi8;  // K bases: hi8, 16+hi8, 32+hi8, 48+hi8
#pragma unroll
    for (int i = 0; i < 8; ++i) {
      cA[g * 8 + i] = cvec[base + i];
      wA[g * 8 + i] = wl[base + i];
    }
  }

  const v16h B1a0 = make_bfrag(Wg1, 32, 0, 0, 64, 32, lane);
  const v16h B1a1 = make_bfrag(Wg1, 32, 32, 0, 64, 32, lane);
  const v16h B1b0 = make_bfrag(Wg1, 32, 0, 16, 64, 32, lane);
  const v16h B1b1 = make_bfrag(Wg1, 32, 32, 16, 64, 32, lane);
  const v16h B2 = make_bfrag(Wg2, 16, 0, 0, 32, 16, lane);
  const v16h B3 = make_bfrag(Wg3, 8, 0, 0, 16, 8, lane);
  const v16h B4 = make_bfrag(Wg4, 2, 0, 0, 8, 2, lane);

  const float b1lo = bg1[nlo];
  const float b1hi = bg1[16 + nlo];
  const float b2 = bg2[nlo];
  const float b3 = (nlo < 8) ? bg3[nlo] : 0.0f;
  const float b4 = (nlo < 2) ? bg4[nlo] : 0.0f;

  float lpsum = 0.0f;
  const int ntiles = (n + 15) >> 4;
  for (int t = blockIdx.x; t < ntiles; t += gridDim.x) {
    const int row = t * 16 + nlo;
    const int rc = row < n ? row : n - 1;
    const float t0 = T[2 * rc];

    // layer 0: g0 = relu(c + t0*wl), directly in A-fragment layout
    v16h a0, a1;
#pragma unroll
    for (int j = 0; j < 16; ++j) {
      const float u = fmaf(t0, wA[j], cA[j]);
      a0[j] = (_Float16)(u > 0.0f ? u : 0.0f);
      const float v = fmaf(t0, wA[16 + j], cA[16 + j]);
      a1[j] = (_Float16)(v > 0.0f ? v : 0.0f);
    }

    // layer 1: 64 -> 32
    v8f acc0, acc1;
#pragma unroll
    for (int r = 0; r < 8; ++r) { acc0[r] = b1lo; acc1[r] = b1hi; }
    acc0 = wmma_f16f32(a0, B1a0, acc0);
    acc0 = wmma_f16f32(a1, B1a1, acc0);
    acc1 = wmma_f16f32(a0, B1b0, acc1);
    acc1 = wmma_f16f32(a1, B1b1, acc1);
    store_relu_half(t1, 40, 0, acc0, lane);
    store_relu_half(t1, 40, 16, acc1, lane);
    __syncthreads();

    // layer 2: 32 -> 16
    const v16h a2 = lds_afrag(t1, 40, 0, lane);
    v8f acc2;
#pragma unroll
    for (int r = 0; r < 8; ++r) acc2[r] = b2;
    acc2 = wmma_f16f32(a2, B2, acc2);
    store_relu_half(t2, 24, 0, acc2, lane);
    __syncthreads();

    // layer 3: 16 -> 8 (K padded 16->32, N padded 8->16; pads are exact zeros)
    const v16h a3 = lds_afrag_k16(t2, 24, lane);
    v8f acc3;
#pragma unroll
    for (int r = 0; r < 8; ++r) acc3[r] = b3;
    acc3 = wmma_f16f32(a3, B3, acc3);
    store_relu_half(t3, 24, 0, acc3, lane);
    __syncthreads();

    // layer 4: 8 -> 2 (K padded 8->32, N padded 2->16), no relu
    const v16h a4 = lds_afrag_k16(t3, 24, lane);
    v8f acc4;
#pragma unroll
    for (int r = 0; r < 8; ++r) acc4[r] = b4;
    acc4 = wmma_f16f32(a4, B4, acc4);

    // stash phi columns in LDS (only lanes of columns 0-1 write; no row-bound guard)
    if (nlo < 2) {
#pragma unroll
      for (int r = 0; r < 8; ++r) phis[2 * (hi8 + r) + nlo] = acc4[r];
    }
    __syncthreads();
    // lanes 0-15: coalesced float2 store of phi row + log-prob accumulation
    if (lane < 16) {
      const int orow = t * 16 + lane;
      if (orow < n) {
        const float mu = phis[2 * lane];
        const float ph1 = phis[2 * lane + 1];
        float2 phv; phv.x = mu; phv.y = ph1;
        *(float2*)(out + 2 * orow) = phv;
        const float y = T[2 * orow + 1];
        const float sig = fmaxf(ph1, 0.0f) + log1pf(expf(-fabsf(ph1)));  // softplus
        const float z = (y - mu) / sig;
        lpsum += -0.91893853320467274178f - logf(sig) - 0.5f * z * z;
      }
    }
    __syncthreads();
  }

  red[lane] = lpsum;
  __syncthreads();
  if (lane == 0) {
    float s = 0.0f;
#pragma unroll
    for (int i = 0; i < 32; ++i) s += red[i];
    atomicAdd(lp_accum, s);
  }
}

// ---------------------------------------------------------------------------
// Kernel 4: finalize mean log-prob.
__global__ void fin_kernel(float* __restrict__ out, const float* __restrict__ ws, int n) {
  if (threadIdx.x == 0) out[2 * n] = ws[32] / (float)n;
}

extern "C" void kernel_launch(void* const* d_in, const int* in_sizes, int n_in,
                              void* d_out, int out_size, void* d_ws, size_t ws_size,
                              hipStream_t stream) {
  const float* O   = (const float*)d_in[0];
  const float* T   = (const float*)d_in[1];
  const float* Wh0 = (const float*)d_in[2];
  const float* bh0 = (const float*)d_in[3];
  const float* Wh1 = (const float*)d_in[4];
  const float* bh1 = (const float*)d_in[5];
  const float* Wh2 = (const float*)d_in[6];
  const float* bh2 = (const float*)d_in[7];
  const float* Wg0 = (const float*)d_in[8];
  const float* bg0 = (const float*)d_in[9];
  const float* Wg1 = (const float*)d_in[10];
  const float* bg1 = (const float*)d_in[11];
  const float* Wg2 = (const float*)d_in[12];
  const float* bg2 = (const float*)d_in[13];
  const float* Wg3 = (const float*)d_in[14];
  const float* bg3 = (const float*)d_in[15];
  const float* Wg4 = (const float*)d_in[16];
  const float* bg4 = (const float*)d_in[17];
  float* out = (float*)d_out;
  float* ws  = (float*)d_ws;

  const int nctx = in_sizes[0] / 2;
  const int ntgt = in_sizes[1] / 2;

  const int enc_tiles = (nctx + 15) >> 4;
  const int dec_tiles = (ntgt + 15) >> 4;
  int enc_blocks = enc_tiles < 3125 ? enc_tiles : 3125;
  int dec_blocks = dec_tiles < 3125 ? dec_tiles : 3125;
  if (enc_blocks < 1) enc_blocks = 1;
  if (dec_blocks < 1) dec_blocks = 1;

  init_ws_kernel<<<1, 64, 0, stream>>>(ws);
  enc_kernel<<<enc_blocks, 32, 0, stream>>>(O, Wh0, bh0, Wh1, bh1, ws, nctx);
  combine_kernel<<<1, 128, 0, stream>>>(Wh2, bh2, Wg0, bg0, ws, nctx);
  dec_kernel<<<dec_blocks, 32, 0, stream>>>(T, Wg0, ws + 64, Wg1, bg1, Wg2, bg2,
                                            Wg3, bg3, Wg4, bg4, out, ws + 32, ntgt);
  fin_kernel<<<1, 1, 0, stream>>>(out, ws, ntgt);
}
